// HierarchicalAttention_55851754717275
// MI455X (gfx1250) — matrix operand order repaired and checked by reading, not compiled, not run on hardware
//
#include <hip/hip_runtime.h>

// MI455X / gfx1250, wave32. f16 WMMA (V_WMMA_F32_16X16X32_F16) with f32 accum.
// Workspace layout (f16): qkv_local[16384,3072] | qkv_global[16384,3072] |
//                         ctx_local[16384,1024] | ctx_global[16384,1024]
// = 268.4 MB total.

typedef _Float16 h8   __attribute__((ext_vector_type(8)));
typedef _Float16 v16h __attribute__((ext_vector_type(16)));
typedef float    v8f  __attribute__((ext_vector_type(8)));
typedef float    f4   __attribute__((ext_vector_type(4)));
typedef int      v4i  __attribute__((ext_vector_type(4)));

// ---- gfx1250 async global->LDS path (guarded; falls back cleanly) ----
// Builtin signature (from hipcc diagnostic): (v4i AS1*, v4i AS3*, imm offset, imm cpol)
#if __has_builtin(__builtin_amdgcn_global_load_async_to_lds_b128) && \
    __has_builtin(__builtin_amdgcn_s_wait_asynccnt)
#define HAVE_ASYNC_LDS 1
#define GPTR(p) ((__attribute__((address_space(1))) v4i*)(p))
#define LPTR(p) ((__attribute__((address_space(3))) v4i*)(p))
#define ASYNC_B128(src, dst) \
  __builtin_amdgcn_global_load_async_to_lds_b128(GPTR(src), LPTR(dst), 0, 0)
#define ASYNC_WAIT() __builtin_amdgcn_s_wait_asynccnt(0)
#else
#define HAVE_ASYNC_LDS 0
#define ASYNC_WAIT()
#endif

__device__ __forceinline__ v8f zero_v8f() {
  v8f z = {0.f,0.f,0.f,0.f,0.f,0.f,0.f,0.f};
  return z;
}

__device__ __forceinline__ v16h combine16(h8 lo, h8 hi) {
  v16h r;
#pragma unroll
  for (int i = 0; i < 8; ++i) { r[i] = lo[i]; r[i+8] = hi[i]; }
  return r;
}

// A fragment: 16-row strip, row-major in LDS (stride ld halves), k window [k0,k0+32)
// ISA layout: lane L -> row M=L&15; elems 0..7 -> K=k0+(L>>4)*8+{0..7};
//             elems 8..15 -> K=k0+16+(L>>4)*8+{0..7}
__device__ __forceinline__ v16h load_a_frag(const _Float16* base, int ld, int k0, int lane) {
  const int m = lane & 15, g = lane >> 4;
  const _Float16* p = base + m*ld + k0 + g*8;
  h8 lo = *(const h8*)(p);
  h8 hi = *(const h8*)(p + 16);
  return combine16(lo, hi);
}

// B fragment from B^T layout [N][K] (stride ld halves):
// lane L -> col N=L&15; elems 0..7 -> K = k0 + (L>>4)*8 + {0..7};
//                       elems 8..15 -> K = k0 + 16 + (L>>4)*8 + {0..7}
__device__ __forceinline__ v16h load_b_frag(const _Float16* base, int ld, int k0, int lane) {
  const int n = lane & 15, g = lane >> 4;
  const _Float16* p = base + n*ld + k0 + g*8;
  h8 lo = *(const h8*)(p);
  h8 hi = *(const h8*)(p + 16);
  return combine16(lo, hi);
}

#define WMMA16(A, B, C) \
  __builtin_amdgcn_wmma_f32_16x16x32_f16(false, (A), false, (B), (short)0, (C), false, false)

// ---------------------------------------------------------------------------
// GEMM with bias: C[M,N] = A[M,K] @ W[N,K]^T + bias[N]
// A is f32 or f16 (A_HALF), C is f16 or f32 (OUT_HALF).
// 128x128 tile per 256-thread block (8 waves as 2x4; wave tile 64 rows x 32 cols).
// ---------------------------------------------------------------------------
template <bool A_HALF, bool OUT_HALF>
__global__ __launch_bounds__(256) void hier_gemm_bias(
    const void* __restrict__ Ap, const float* __restrict__ W,
    const float* __restrict__ bias, void* __restrict__ Cp,
    int M, int N, int K)
{
  constexpr int BK = 32;
  constexpr int LD = BK + 8;   // 40 halves = 80 B rows: 16B-aligned + bank skew
  __shared__ __align__(16) _Float16 As[128 * LD];
  __shared__ __align__(16) _Float16 Bs[128 * LD];

  const int tid  = threadIdx.x;
  const int lane = tid & 31;
  const int wave = tid >> 5;
  const int wr   = wave >> 2;            // 0..1 : 64-row strip
  const int wc   = wave & 3;             // 0..3 : 32-col strip
  const int row0 = blockIdx.x * 128;
  const int col0 = blockIdx.y * 128;

  const int sr = tid >> 1;               // staging row 0..127
  const int sk = (tid & 1) * 16;         // staging k offset {0,16}

  v8f acc[4][2];
#pragma unroll
  for (int i = 0; i < 4; ++i) { acc[i][0] = zero_v8f(); acc[i][1] = zero_v8f(); }

  for (int k0 = 0; k0 < K; k0 += BK) {
    // ---- stage A tile ----
    if constexpr (A_HALF) {
      const _Float16* A = (const _Float16*)Ap;
      const _Float16* src = A + (size_t)(row0 + sr) * K + k0 + sk;
#if HAVE_ASYNC_LDS
      ASYNC_B128(src,     &As[sr*LD + sk]);       // 8 halves = 16 B per lane
      ASYNC_B128(src + 8, &As[sr*LD + sk + 8]);
#else
      *(h8*)(&As[sr*LD + sk])     = *(const h8*)(src);
      *(h8*)(&As[sr*LD + sk + 8]) = *(const h8*)(src + 8);
#endif
      if (k0 + BK < K) __builtin_prefetch((const void*)(src + BK), 0, 1);
    } else {
      const float* A = (const float*)Ap;
      const float* src = A + (size_t)(row0 + sr) * K + k0 + sk;
      f4 x0 = ((const f4*)src)[0], x1 = ((const f4*)src)[1];
      f4 x2 = ((const f4*)src)[2], x3 = ((const f4*)src)[3];
      h8 y0, y1;
#pragma unroll
      for (int i = 0; i < 4; ++i) {
        y0[i] = (_Float16)x0[i]; y0[i+4] = (_Float16)x1[i];
        y1[i] = (_Float16)x2[i]; y1[i+4] = (_Float16)x3[i];
      }
      *(h8*)(&As[sr*LD + sk])     = y0;
      *(h8*)(&As[sr*LD + sk + 8]) = y1;
      if (k0 + BK < K) __builtin_prefetch((const void*)(src + BK), 0, 1);
    }
    // ---- stage B tile: W rows are already [N][K] ("B^T") layout; f32->f16 ----
    {
      const float* src = W + (size_t)(col0 + sr) * K + k0 + sk;
      f4 x0 = ((const f4*)src)[0], x1 = ((const f4*)src)[1];
      f4 x2 = ((const f4*)src)[2], x3 = ((const f4*)src)[3];
      h8 y0, y1;
#pragma unroll
      for (int i = 0; i < 4; ++i) {
        y0[i] = (_Float16)x0[i]; y0[i+4] = (_Float16)x1[i];
        y1[i] = (_Float16)x2[i]; y1[i+4] = (_Float16)x3[i];
      }
      *(h8*)(&Bs[sr*LD + sk])     = y0;
      *(h8*)(&Bs[sr*LD + sk + 8]) = y1;
      if (k0 + BK < K) __builtin_prefetch((const void*)(src + BK), 0, 1);
    }
    if constexpr (A_HALF) { ASYNC_WAIT(); }
    __syncthreads();

    v16h af[4], bf[2];
#pragma unroll
    for (int rt = 0; rt < 4; ++rt)
      af[rt] = load_a_frag(&As[(wr*64 + rt*16) * LD], LD, 0, lane);
#pragma unroll
    for (int ct = 0; ct < 2; ++ct)
      bf[ct] = load_b_frag(&Bs[(wc*32 + ct*16) * LD], LD, 0, lane);
#pragma unroll
    for (int rt = 0; rt < 4; ++rt)
#pragma unroll
      for (int ct = 0; ct < 2; ++ct)
        acc[rt][ct] = WMMA16(af[rt], bf[ct], acc[rt][ct]);
    __syncthreads();
  }

  // ---- epilogue: bias add + store (C layout: row = rt*16 + (lane>>4)*8 + r) ----
  const int g = lane >> 4, nl = lane & 15;
#pragma unroll
  for (int ct = 0; ct < 2; ++ct) {
    const int col = col0 + wc*32 + ct*16 + nl;
    const float bv = bias[col];
#pragma unroll
    for (int rt = 0; rt < 4; ++rt) {
#pragma unroll
      for (int r = 0; r < 8; ++r) {
        const int row = row0 + wr*64 + rt*16 + g*8 + r;
        const float v = acc[rt][ct][r] + bv;
        if constexpr (OUT_HALF) ((_Float16*)Cp)[(size_t)row * N + col] = (_Float16)v;
        else                    ((float*)Cp)[(size_t)row * N + col]    = v;
      }
    }
  }
}

// ---------------------------------------------------------------------------
// Flash-style windowed attention. One block per (128-row q-block, head).
// qkv: [16384, 3072] f16 (q | k | v, each with head-major feature h*64+e).
// ctx: [16384, 1024] f16. Reference semantics: scores = qk/8 + (+1.0 strictly
// above the window diagonal), full softmax over the window.
// ---------------------------------------------------------------------------
__global__ __launch_bounds__(256) void hier_flash_attn(
    const _Float16* __restrict__ qkv, _Float16* __restrict__ ctx, int window)
{
  constexpr int DH = 64, BQ = 128, BKV = 64;
  constexpr int LD = 72;  // 144 B rows: 16B-aligned + bank skew
  __shared__ __align__(16) _Float16 Qs[BQ  * LD];
  __shared__ __align__(16) _Float16 Ks[BKV * LD];
  __shared__ __align__(16) _Float16 Vt[DH  * LD];  // V transposed: [dh][key]
  __shared__ __align__(16) _Float16 Ps[BQ  * LD];

  const int tid  = threadIdx.x;
  const int lane = tid & 31;
  const int wave = tid >> 5;                // 8 waves -> 16-row strips
  const int h    = blockIdx.y;
  const int t0   = blockIdx.x * BQ;         // first q token of this block
  const int wstart = (t0 / window) * window;

  // ---- stage Q: 128 rows x 64 halves ----
  {
    const int r = tid >> 1, c0 = (tid & 1) * 32;
    const _Float16* src = qkv + (size_t)(t0 + r) * 3072 + h*DH + c0;
    _Float16* dst = &Qs[r*LD + c0];
#if HAVE_ASYNC_LDS
#pragma unroll
    for (int i = 0; i < 4; ++i) ASYNC_B128(src + i*8, dst + i*8);
    ASYNC_WAIT();
#else
#pragma unroll
    for (int i = 0; i < 4; ++i)
      *(h8*)(dst + i*8) = *(const h8*)(src + i*8);
#endif
  }
  __syncthreads();

  v16h qf[2];
#pragma unroll
  for (int ks = 0; ks < 2; ++ks)
    qf[ks] = load_a_frag(&Qs[(wave*16) * LD], LD, ks*32, lane);

  const int g = lane >> 4, nl = lane & 15;
  float mrow[8], lrow[8];
#pragma unroll
  for (int r = 0; r < 8; ++r) { mrow[r] = -1e30f; lrow[r] = 0.f; }
  v8f O[4];
#pragma unroll
  for (int nt = 0; nt < 4; ++nt) O[nt] = zero_v8f();

  for (int kb = wstart; kb < wstart + window; kb += BKV) {
    __syncthreads();  // previous iteration's Ks/Vt/Ps reads complete
    // ---- stage K block: [key][dh] (already the [N][K] layout for S=Q@K^T) ----
    {
      const int r = tid >> 2, c0 = (tid & 3) * 16;
      const _Float16* src = qkv + (size_t)(kb + r) * 3072 + 1024 + h*DH + c0;
#if HAVE_ASYNC_LDS
      ASYNC_B128(src,     &Ks[r*LD + c0]);
      ASYNC_B128(src + 8, &Ks[r*LD + c0 + 8]);
#else
      *(h8*)(&Ks[r*LD + c0])     = *(const h8*)(src);
      *(h8*)(&Ks[r*LD + c0 + 8]) = *(const h8*)(src + 8);
#endif
    }
    // ---- stage V transposed: Vt[dh][key] = [N][K] layout for O=P@V ----
    {
      const int key = tid >> 2, e0 = (tid & 3) * 16;
      const _Float16* src = qkv + (size_t)(kb + key) * 3072 + 2048 + h*DH + e0;
      h8 a = *(const h8*)(src);
      h8 b = *(const h8*)(src + 8);
#pragma unroll
      for (int i = 0; i < 8; ++i) {
        Vt[(e0 + i) * LD + key]     = a[i];
        Vt[(e0 + 8 + i) * LD + key] = b[i];
      }
    }
    ASYNC_WAIT();
    __syncthreads();

    // ---- S strip (16 x 64) = Q @ K^T ----
    v8f s[4];
#pragma unroll
    for (int nt = 0; nt < 4; ++nt) s[nt] = zero_v8f();
#pragma unroll
    for (int ks = 0; ks < 2; ++ks) {
#pragma unroll
      for (int nt = 0; nt < 4; ++nt) {
        v16h kf = load_b_frag(&Ks[(nt*16) * LD], LD, ks*32, lane);
        s[nt] = WMMA16(qf[ks], kf, s[nt]);
      }
    }

    // ---- scale 1/sqrt(64) + reference's additive upper-triangular +1.0 ----
    const int qbase = t0 + wave*16 + g*8;
#pragma unroll
    for (int nt = 0; nt < 4; ++nt) {
      const int kt = kb + nt*16 + nl;
#pragma unroll
      for (int r = 0; r < 8; ++r) {
        float x = s[nt][r] * 0.125f;
        if (kt > qbase + r) x += 1.0f;
        s[nt][r] = x;
      }
    }

    // ---- online softmax; each row lives in one 16-lane half-wave ----
#pragma unroll
    for (int r = 0; r < 8; ++r) {
      float mb = s[0][r];
#pragma unroll
      for (int nt = 1; nt < 4; ++nt) mb = fmaxf(mb, s[nt][r]);
#pragma unroll
      for (int off = 1; off < 16; off <<= 1) mb = fmaxf(mb, __shfl_xor(mb, off, 16));
      const float mn    = fmaxf(mrow[r], mb);
      const float alpha = __expf(mrow[r] - mn);
      float sum = 0.f;
#pragma unroll
      for (int nt = 0; nt < 4; ++nt) {
        float p = __expf(s[nt][r] - mn);
        s[nt][r] = p;
        sum += p;
      }
#pragma unroll
      for (int off = 1; off < 16; off <<= 1) sum += __shfl_xor(sum, off, 16);
      lrow[r] = lrow[r] * alpha + sum;
      mrow[r] = mn;
#pragma unroll
      for (int nt = 0; nt < 4; ++nt) O[nt][r] *= alpha;
    }

    // ---- spill P to LDS (f16) so it can re-enter as a WMMA A operand ----
#pragma unroll
    for (int nt = 0; nt < 4; ++nt)
#pragma unroll
      for (int r = 0; r < 8; ++r)
        Ps[(wave*16 + g*8 + r) * LD + nt*16 + nl] = (_Float16)s[nt][r];
    __syncthreads();

    // ---- O += P @ V ----
#pragma unroll
    for (int ks = 0; ks < 2; ++ks) {
      v16h pf = load_a_frag(&Ps[(wave*16) * LD], LD, ks*32, lane);
#pragma unroll
      for (int nt = 0; nt < 4; ++nt) {
        v16h vf = load_b_frag(&Vt[(nt*16) * LD], LD, ks*32, lane);
        O[nt] = WMMA16(pf, vf, O[nt]);
      }
    }
  }

  // ---- finalize: divide by softmax denominator, write ctx f16 ----
#pragma unroll
  for (int r = 0; r < 8; ++r) {
    const float inv = 1.0f / lrow[r];
    const int t = t0 + wave*16 + g*8 + r;
#pragma unroll
    for (int nt = 0; nt < 4; ++nt)
      ctx[(size_t)t * 1024 + h*DH + nt*16 + nl] = (_Float16)(O[nt][r] * inv);
  }
}

// ---------------------------------------------------------------------------
extern "C" void kernel_launch(void* const* d_in, const int* in_sizes, int n_in,
                              void* d_out, int out_size, void* d_ws, size_t ws_size,
                              hipStream_t stream) {
  (void)in_sizes; (void)n_in; (void)out_size; (void)ws_size;
  const float* lqs    = (const float*)d_in[0];
  const float* gqs    = (const float*)d_in[1];
  const float* wl_in  = (const float*)d_in[2];
  const float* bl_in  = (const float*)d_in[3];
  const float* wl_out = (const float*)d_in[4];
  const float* bl_out = (const float*)d_in[5];
  const float* wg_in  = (const float*)d_in[6];
  const float* bg_in  = (const float*)d_in[7];
  const float* wg_out = (const float*)d_in[8];
  const float* bg_out = (const float*)d_in[9];
  float* out = (float*)d_out;

  const size_t T = (size_t)4 * 4096;          // 16384 tokens per variant
  _Float16* qkv_l = (_Float16*)d_ws;
  _Float16* qkv_g = qkv_l + T * 3072;
  _Float16* ctx_l = qkv_g + T * 3072;
  _Float16* ctx_g = ctx_l + T * 1024;         // total ws: 268.4 MB

  dim3 blk(256);
  // QKV projections (f32 in -> f16 qkv)
  hier_gemm_bias<false, true><<<dim3(128, 24), blk, 0, stream>>>(
      (const void*)lqs, wl_in, bl_in, (void*)qkv_l, 16384, 3072, 1024);
  hier_gemm_bias<false, true><<<dim3(128, 24), blk, 0, stream>>>(
      (const void*)gqs, wg_in, bg_in, (void*)qkv_g, 16384, 3072, 1024);
  // Windowed attention (128 q-blocks x 16 heads)
  hier_flash_attn<<<dim3(128, 16), blk, 0, stream>>>(qkv_l, ctx_l, 128);
  hier_flash_attn<<<dim3(128, 16), blk, 0, stream>>>(qkv_g, ctx_g, 1024);
  // Output projections (f16 ctx -> f32 d_out)
  hier_gemm_bias<true, false><<<dim3(128, 8), blk, 0, stream>>>(
      (const void*)ctx_l, wl_out, bl_out, (void*)out, 16384, 1024, 1024);
  hier_gemm_bias<true, false><<<dim3(128, 8), blk, 0, stream>>>(
      (const void*)ctx_g, wg_out, bg_out, (void*)(out + T * 1024), 16384, 1024, 1024);
}
